// SAGE_24386824306871
// MI455X (gfx1250) — compile-verified, matrix operand order checked by reading, not run on hardware
//
#include <hip/hip_runtime.h>

#define N0 200000
#define N1 50000
#define N2 12500
#define N3 3125
#define E1 800000
#define E2 200000
#define E3 50000
#define DIN 128
#define DH  256
#define DOUT 128

typedef __attribute__((ext_vector_type(16))) __bf16 v16bf;
typedef __attribute__((ext_vector_type(8)))  __bf16 v8bf;
typedef __attribute__((ext_vector_type(8)))  float  v8f;

// -------------------- degree counts --------------------
__global__ void deg_kernel(const int* __restrict__ dst, float* __restrict__ cnt, int E) {
  int e = blockIdx.x * blockDim.x + threadIdx.x;
  if (e < E) atomicAdd(&cnt[dst[e]], 1.0f);
}

// -------------------- edge scatter-add (segment sum) --------------------
// one thread per (edge, float4 chunk): coalesced gather, 4 f32 atomics (L2-resident accumulator)
__global__ void scatter_kernel(const float* __restrict__ x,
                               const int* __restrict__ src,
                               const int* __restrict__ dst,
                               float* __restrict__ aggr, int E, int K) {
  const int per = K >> 2;
  const int idx = blockIdx.x * blockDim.x + threadIdx.x;
  if (idx >= E * per) return;
  const int e = idx / per;
  const int c = idx - e * per;
  const float4 v = ((const float4*)(x + (size_t)src[e] * K))[c];
  float* d = aggr + (size_t)dst[e] * K + (c << 2);
  atomicAdd(d + 0, v.x);
  atomicAdd(d + 1, v.y);
  atomicAdd(d + 2, v.z);
  atomicAdd(d + 3, v.w);
}

// -------------------- f32 -> bf16 converts (hoisted out of GEMM) --------------------
// rows >= M are zero-filled padding (so WMMA fragments need no masking)
__global__ void cvt_rows_kernel(const float* __restrict__ in, __bf16* __restrict__ out,
                                int M, int Mpad, int kshift) {
  const int i = blockIdx.x * blockDim.x + threadIdx.x;
  if (i >= (Mpad << kshift)) return;
  const int m = i >> kshift;
  out[i] = (m < M) ? (__bf16)in[i] : (__bf16)0.0f;
}

// fused mean-normalize + convert for the aggregated messages
__global__ void norm_cvt_kernel(const float* __restrict__ aggr, const float* __restrict__ cnt,
                                __bf16* __restrict__ out, int M, int Mpad, int kshift) {
  const int i = blockIdx.x * blockDim.x + threadIdx.x;
  if (i >= (Mpad << kshift)) return;
  const int m = i >> kshift;
  float v = 0.0f;
  if (m < M) v = aggr[i] / fmaxf(cnt[m], 1.0f);
  out[i] = (__bf16)v;
}

// -------------------- WMMA GEMM --------------------
// 16-bit A/B fragment K placement (ISA 7.12.2): lane half h = lane>>4,
// elements 0..7  <- K = k0 + 8h .. +7   (16 contiguous bytes)
// elements 8..15 <- K = k0 + 16 + 8h .. +7 (16 contiguous bytes)
__device__ __forceinline__ v16bf load_frag(const __bf16* __restrict__ p, int h) {
  const v8bf lo = *(const v8bf*)(p + (h << 3));
  const v8bf hi = *(const v8bf*)(p + 16 + (h << 3));
  return __builtin_shufflevector(lo, hi, 0, 1, 2, 3, 4, 5, 6, 7,
                                 8, 9, 10, 11, 12, 13, 14, 15);
}

// out[m,n] = relu( meanaggr[m,:] . Wl[n,:] + xdst[m,:] . Wr[n,:] + bl[n] )
// one wave per 16x64 output strip: 1 A fragment feeds 4 WMMAs per k-step
__global__ void sage_gemm_kernel(const __bf16* __restrict__ Ab,  // Mpad x K (normalized aggr)
                                 const __bf16* __restrict__ Xb,  // Mpad x K (dst features)
                                 const __bf16* __restrict__ Wlb, // N x K
                                 const __bf16* __restrict__ Wrb, // N x K
                                 const float* __restrict__ bl,
                                 float* __restrict__ out,        // M x N
                                 int M, int N, int K, int relu) {
  const int lane = threadIdx.x & 31;
  const int wave = threadIdx.x >> 5;
  const int grp  = blockIdx.x * (blockDim.x >> 5) + wave;
  const int ngrp = N >> 6;                 // 64-wide column groups
  const int mtiles = (M + 15) >> 4;
  if (grp >= mtiles * ngrp) return;
  const int mt = grp / ngrp;
  const int ng = grp - mt * ngrp;
  const int m0 = mt << 4;
  const int n0 = ng << 6;
  const int h = lane >> 4;
  const int l16 = lane & 15;

  const __bf16* arow = Ab + (size_t)(m0 + l16) * K;   // padded: always in-bounds
  const __bf16* xrow = Xb + (size_t)(m0 + l16) * K;
  const __bf16* wl0 = Wlb + (size_t)(n0 + l16) * K;
  const __bf16* wr0 = Wrb + (size_t)(n0 + l16) * K;
  const size_t wstride = (size_t)16 * K;

  v8f c0 = {}, c1 = {}, c2 = {}, c3 = {};

  for (int k0 = 0; k0 < K; k0 += 32) {
    // pass 1: mean-aggregated neighbors x Wl^T
    v16bf a = load_frag(arow + k0, h);
    v16bf b0 = load_frag(wl0 + k0, h);
    v16bf b1 = load_frag(wl0 + wstride + k0, h);
    v16bf b2 = load_frag(wl0 + 2 * wstride + k0, h);
    v16bf b3 = load_frag(wl0 + 3 * wstride + k0, h);
    c0 = __builtin_amdgcn_wmma_f32_16x16x32_bf16(false, a, false, b0, (short)0, c0, false, false);
    c1 = __builtin_amdgcn_wmma_f32_16x16x32_bf16(false, a, false, b1, (short)0, c1, false, false);
    c2 = __builtin_amdgcn_wmma_f32_16x16x32_bf16(false, a, false, b2, (short)0, c2, false, false);
    c3 = __builtin_amdgcn_wmma_f32_16x16x32_bf16(false, a, false, b3, (short)0, c3, false, false);
    // pass 2: root (dst) features x Wr^T
    v16bf x = load_frag(xrow + k0, h);
    v16bf d0 = load_frag(wr0 + k0, h);
    v16bf d1 = load_frag(wr0 + wstride + k0, h);
    v16bf d2 = load_frag(wr0 + 2 * wstride + k0, h);
    v16bf d3 = load_frag(wr0 + 3 * wstride + k0, h);
    c0 = __builtin_amdgcn_wmma_f32_16x16x32_bf16(false, x, false, d0, (short)0, c0, false, false);
    c1 = __builtin_amdgcn_wmma_f32_16x16x32_bf16(false, x, false, d1, (short)0, c1, false, false);
    c2 = __builtin_amdgcn_wmma_f32_16x16x32_bf16(false, x, false, d2, (short)0, c2, false, false);
    c3 = __builtin_amdgcn_wmma_f32_16x16x32_bf16(false, x, false, d3, (short)0, c3, false, false);
  }

  // C/D layout: element v of lane L -> m = m0 + v + 8*(L>>4), n = ntile + (L&15)
  v8f acc[4] = {c0, c1, c2, c3};
#pragma unroll
  for (int t = 0; t < 4; ++t) {
    const int n = n0 + t * 16 + l16;
    const float bias = bl[n];
#pragma unroll
    for (int v = 0; v < 8; ++v) {
      const int m = m0 + v + (h << 3);
      float r = acc[t][v] + bias;
      if (relu) r = fmaxf(r, 0.0f);
      if (m < M) out[(size_t)m * N + n] = r;
    }
  }
}

extern "C" void kernel_launch(void* const* d_in, const int* in_sizes, int n_in,
                              void* d_out, int out_size, void* d_ws, size_t ws_size,
                              hipStream_t stream) {
  const float* x   = (const float*)d_in[0];
  const int*   ei1 = (const int*)d_in[1];
  const int*   ei2 = (const int*)d_in[2];
  const int*   ei3 = (const int*)d_in[3];
  const float* Wl1 = (const float*)d_in[4];
  const float* bl1 = (const float*)d_in[5];
  const float* Wr1 = (const float*)d_in[6];
  const float* Wl2 = (const float*)d_in[7];
  const float* bl2 = (const float*)d_in[8];
  const float* Wr2 = (const float*)d_in[9];
  const float* Wl3 = (const float*)d_in[10];
  const float* bl3 = (const float*)d_in[11];
  const float* Wr3 = (const float*)d_in[12];

  float* h1 = (float*)d_out;                 // N1 x DH
  float* h2 = h1 + (size_t)N1 * DH;          // N2 x DH
  float* h3 = h2 + (size_t)N2 * DH;          // N3 x DOUT

  const int M2p = ((N2 + 15) / 16) * 16;     // 12512
  const int M3p = ((N3 + 15) / 16) * 16;     // 3136

  // ---- workspace layout: f32 accumulators, then 32B-aligned bf16 region ----
  float* ws = (float*)d_ws;
  size_t o = 0;
  float* aggr1 = ws + o; o += (size_t)N1 * DIN;
  float* cnt1  = ws + o; o += N1;
  float* aggr2 = ws + o; o += (size_t)N2 * DH;
  float* cnt2  = ws + o; o += N2;
  float* aggr3 = ws + o; o += (size_t)N3 * DH;
  float* cnt3  = ws + o; o += N3;
  o = (o + 7) & ~(size_t)7;                  // 32B-align the bf16 region
  hipMemsetAsync(d_ws, 0, o * sizeof(float), stream);  // zero accumulators (capture legal)

  __bf16* bws = (__bf16*)(ws + o);
  size_t b = 0;
  __bf16* wl1b = bws + b; b += (size_t)DH * DIN;
  __bf16* wr1b = bws + b; b += (size_t)DH * DIN;
  __bf16* wl2b = bws + b; b += (size_t)DH * DH;
  __bf16* wr2b = bws + b; b += (size_t)DH * DH;
  __bf16* wl3b = bws + b; b += (size_t)DOUT * DH;
  __bf16* wr3b = bws + b; b += (size_t)DOUT * DH;
  __bf16* a1b  = bws + b; b += (size_t)N1 * DIN;
  __bf16* x1b  = bws + b; b += (size_t)N1 * DIN;
  __bf16* a2b  = bws + b; b += (size_t)M2p * DH;
  __bf16* x2b  = bws + b; b += (size_t)M2p * DH;
  __bf16* a3b  = bws + b; b += (size_t)M3p * DH;
  __bf16* x3b  = bws + b; b += (size_t)M3p * DH;

  // ---- weight conversions (once per launch; tiny) ----
  cvt_rows_kernel<<<(DH * DIN + 255) / 256, 256, 0, stream>>>(Wl1, wl1b, DH, DH, 7);
  cvt_rows_kernel<<<(DH * DIN + 255) / 256, 256, 0, stream>>>(Wr1, wr1b, DH, DH, 7);
  cvt_rows_kernel<<<(DH * DH + 255) / 256, 256, 0, stream>>>(Wl2, wl2b, DH, DH, 8);
  cvt_rows_kernel<<<(DH * DH + 255) / 256, 256, 0, stream>>>(Wr2, wr2b, DH, DH, 8);
  cvt_rows_kernel<<<(DOUT * DH + 255) / 256, 256, 0, stream>>>(Wl3, wl3b, DOUT, DOUT, 8);
  cvt_rows_kernel<<<(DOUT * DH + 255) / 256, 256, 0, stream>>>(Wr3, wr3b, DOUT, DOUT, 8);

  // -------- layer 1: x (N0 x DIN) -> h1 (N1 x DH), ReLU --------
  {
    const int *src = ei1, *dst = ei1 + E1;
    deg_kernel<<<(E1 + 255) / 256, 256, 0, stream>>>(dst, cnt1, E1);
    const int total = E1 * (DIN / 4);
    scatter_kernel<<<(total + 255) / 256, 256, 0, stream>>>(x, src, dst, aggr1, E1, DIN);
    const int ne = N1 * DIN;
    norm_cvt_kernel<<<(ne + 255) / 256, 256, 0, stream>>>(aggr1, cnt1, a1b, N1, N1, 7);
    cvt_rows_kernel<<<(ne + 255) / 256, 256, 0, stream>>>(x, x1b, N1, N1, 7);
    const int grps = ((N1 + 15) / 16) * (DH / 64);
    sage_gemm_kernel<<<(grps + 7) / 8, 256, 0, stream>>>(a1b, x1b, wl1b, wr1b, bl1,
                                                         h1, N1, DH, DIN, 1);
  }
  // -------- layer 2: h1 -> h2 (N2 x DH), ReLU --------
  {
    const int *src = ei2, *dst = ei2 + E2;
    deg_kernel<<<(E2 + 255) / 256, 256, 0, stream>>>(dst, cnt2, E2);
    const int total = E2 * (DH / 4);
    scatter_kernel<<<(total + 255) / 256, 256, 0, stream>>>(h1, src, dst, aggr2, E2, DH);
    const int ne = M2p * DH;
    norm_cvt_kernel<<<(ne + 255) / 256, 256, 0, stream>>>(aggr2, cnt2, a2b, N2, M2p, 8);
    cvt_rows_kernel<<<(ne + 255) / 256, 256, 0, stream>>>(h1, x2b, N2, M2p, 8);
    const int grps = ((N2 + 15) / 16) * (DH / 64);
    sage_gemm_kernel<<<(grps + 7) / 8, 256, 0, stream>>>(a2b, x2b, wl2b, wr2b, bl2,
                                                         h2, N2, DH, DH, 1);
  }
  // -------- layer 3: h2 -> h3 (N3 x DOUT), no ReLU --------
  {
    const int *src = ei3, *dst = ei3 + E3;
    deg_kernel<<<(E3 + 255) / 256, 256, 0, stream>>>(dst, cnt3, E3);
    const int total = E3 * (DH / 4);
    scatter_kernel<<<(total + 255) / 256, 256, 0, stream>>>(h2, src, dst, aggr3, E3, DH);
    const int ne = M3p * DH;
    norm_cvt_kernel<<<(ne + 255) / 256, 256, 0, stream>>>(aggr3, cnt3, a3b, N3, M3p, 8);
    cvt_rows_kernel<<<(ne + 255) / 256, 256, 0, stream>>>(h2, x3b, N3, M3p, 8);
    const int grps = ((N3 + 15) / 16) * (DOUT / 64);
    sage_gemm_kernel<<<(grps + 7) / 8, 256, 0, stream>>>(a3b, x3b, wl3b, wr3b, bl3,
                                                         h3, N3, DOUT, DH, 0);
  }

  (void)in_sizes; (void)n_in; (void)out_size; (void)ws_size;
}
